// SpaFEM_23192823398809
// MI455X (gfx1250) — matrix operand order repaired and checked by reading, not compiled, not run on hardware
//
#include <hip/hip_runtime.h>
#include <math.h>

// ---------------- problem constants ----------------
#define CC   96
#define HH   128
#define WW_  128
#define HW   16384      // 128*128
#define PP   4
#define EE   192
#define DI   384
#define NSZ  16
#define RRk  12
#define KDIR 4
#define NBLK 8
#define GWC  48
#define CPP  1536       // C*P*P
#define HP   32
#define LL   1024       // HP*WP

typedef __attribute__((ext_vector_type(16))) _Float16 v16h;
typedef __attribute__((ext_vector_type(8)))  float    v8f;
typedef __attribute__((ext_vector_type(4)))  unsigned v4u;
typedef __attribute__((ext_vector_type(4)))  int      v4i;
typedef __attribute__((ext_vector_type(8)))  int      v8i_;

union PackU { unsigned u; _Float16 h[2]; };

// =====================================================================
// Fragment packing: f32 matrix -> WMMA f16 fragment layout.
// A-frag (16x32, ISA 7.12.2): lane m=lane&15, half=lane>>4,
//   vgpr v holds K pair at kb = (v%4)*2 + (v/4)*16 + half*8.
// Fragment = 256 dwords, frag index = mt*Kt + kt.
// =====================================================================
__global__ void k_pack_a(const float* __restrict__ src, unsigned* __restrict__ dst,
                         int M, int Kd, int ld, int Kt) {
  int tid  = blockIdx.x * 256 + threadIdx.x;
  int frag = tid >> 8, w = tid & 255;
  int lane = w >> 3, v = w & 7;
  int mt = frag / Kt, kt = frag - mt * Kt;
  int m  = mt * 16 + (lane & 15);
  int kb = ((v & 3) << 1) + ((v >> 2) << 4) + ((lane >> 4) << 3);
  int k  = kt * 32 + kb;
  float f0 = 0.f, f1 = 0.f;
  if (m < M) {
    if (k     < Kd) f0 = src[(size_t)m * ld + k];
    if (k + 1 < Kd) f1 = src[(size_t)m * ld + k + 1];
  }
  PackU p; p.h[0] = (_Float16)f0; p.h[1] = (_Float16)f1;
  dst[tid] = p.u;
}

// B-frag (32x16): lane n=lane&15, half=lane>>4, vgpr v holds K pair at
//   kb = (v%4)*2 + (v/4)*16 + half*8.  trans!=0 -> src is [N][Kd] row-major (weight.T use).
__global__ void k_pack_b(const float* __restrict__ src, unsigned* __restrict__ dst,
                         int Kd, int N, int ld, int Kt, int trans) {
  int tid  = blockIdx.x * 256 + threadIdx.x;
  int frag = tid >> 8, w = tid & 255;
  int lane = w >> 3, v = w & 7;
  int nt = frag / Kt, kt = frag - nt * Kt;
  int n  = nt * 16 + (lane & 15);
  int kb = ((v & 3) << 1) + ((v >> 2) << 4) + ((lane >> 4) << 3);
  int k  = kt * 32 + kb;
  float f0 = 0.f, f1 = 0.f;
  if (n < N) {
    if (trans) {
      if (k     < Kd) f0 = src[(size_t)n * ld + k];
      if (k + 1 < Kd) f1 = src[(size_t)n * ld + k + 1];
    } else {
      if (k     < Kd) f0 = src[(size_t)k * ld + n];
      if (k + 1 < Kd) f1 = src[(size_t)(k + 1) * ld + n];
    }
  }
  PackU p; p.h[0] = (_Float16)f0; p.h[1] = (_Float16)f1;
  dst[tid] = p.u;
}

// --------------- shared epilogue for both GEMM kernels ---------------
__device__ __forceinline__ void gemm_epilogue(v8f acc, float* __restrict__ C,
                                              int mt, int nt, int lane, int ldc,
                                              const float* __restrict__ bias, int bias_mode,
                                              int act, const float* __restrict__ resid) {
  // C/D layout: lanes 0-15: M=r, N=lane; lanes 16-31: M=8+r, N=lane-16
  int n    = nt * 16 + (lane & 15);
  int mrow = mt * 16 + ((lane >> 4) << 3);
#pragma unroll
  for (int r = 0; r < 8; ++r) {
    int m = mrow + r;
    float val = acc[r];
    if (bias_mode == 1)      val += bias[n];
    else if (bias_mode == 2) val += bias[m];
    if (resid) val += resid[(size_t)m * ldc + n];
    if (act == 1)      val = val >= 0.f ? val : 0.2f * val;
    else if (act == 2) val = 1.f / (1.f + __expf(-val));
    else if (act == 3) val = (val > 20.f) ? val : log1pf(__expf(val));
    else if (act == 4) val = val / (1.f + __expf(-val));
    C[(size_t)m * ldc + n] = val;
  }
}

// =====================================================================
// Direct-global wave-level WMMA GEMM (small / odd-shaped cases).
// =====================================================================
__global__ void k_gemm(const v16h* __restrict__ Ap, const v16h* __restrict__ Bp,
                       float* __restrict__ C, int Kt, int ldc,
                       const float* __restrict__ bias, int bias_mode,
                       int act, const float* __restrict__ resid) {
  int nt = blockIdx.x, mt = blockIdx.y;
  int lane = threadIdx.x;
  const v16h* aP = Ap + ((size_t)mt * Kt) * 32 + lane;   // 32 v16h per fragment
  const v16h* bP = Bp + ((size_t)nt * Kt) * 32 + lane;
  v8f c0 = {}; v8f c1 = {};
  int pairs = Kt >> 1;
  for (int i = 0; i < pairs; ++i) {
    v16h a0 = aP[0],  b0 = bP[0];
    v16h a1 = aP[32], b1 = bP[32];
    __builtin_prefetch((const void*)(aP + 64), 0, 3);
    __builtin_prefetch((const void*)(bP + 64), 0, 3);
    c0 = __builtin_amdgcn_wmma_f32_16x16x32_f16(false, a0, false, b0, (short)0, c0, false, false);
    c1 = __builtin_amdgcn_wmma_f32_16x16x32_f16(false, a1, false, b1, (short)0, c1, false, false);
    aP += 64; bP += 64;
  }
  if (Kt & 1) {
    v16h a0 = aP[0], b0 = bP[0];
    c0 = __builtin_amdgcn_wmma_f32_16x16x32_f16(false, a0, false, b0, (short)0, c0, false, false);
  }
  gemm_epilogue(c0 + c1, C, mt, nt, lane, ldc, bias, bias_mode, act, resid);
}

// =====================================================================
// Tensor-Data-Mover staged GEMM: block = 4 waves = 4 M-tiles sharing one
// N-tile.  Wave 0 issues two TDM DMAs (A strips, B strip) into LDS using
// a 1-D D# descriptor (ISA ch.8), waits on TENSORcnt, then all 4 waves
// run the WMMA loop out of LDS (B strip reused 4x).
// =====================================================================
__device__ __forceinline__ void tdm_load_1d(const void* gptr, unsigned lds_byte_off,
                                            unsigned n_dwords) {
  unsigned long long ga = (unsigned long long)(uintptr_t)gptr;
  v4u g0;
  g0.x = 1u;                                         // count=1, user descriptor
  g0.y = lds_byte_off;                               // lds_addr
  g0.z = (unsigned)ga;                               // global_addr[31:0]
  g0.w = (unsigned)((ga >> 32) & 0x1FFFFFFu) | (2u << 30);  // addr[56:32] | type=2
  v8i_ g1;
  g1[0] = (int)(2u << 16);                           // data_size = 4 bytes
  g1[1] = (int)((n_dwords & 0xFFFFu) << 16);         // tensor_dim0[15:0]
  g1[2] = (int)((n_dwords >> 16) & 0xFFFFu);         // tensor_dim0[31:16], tensor_dim1 lo = 0
  g1[3] = (int)((n_dwords & 0xFFFFu) << 16);         // tile_dim0 (<=65535)
  g1[4] = 0;                                         // tile_dim1=0, tile_dim2=0 (1-D tile)
  g1[5] = (int)n_dwords;                             // tensor_dim0_stride[31:0]
  g1[6] = 0;
  g1[7] = 0;
  v4i z4 = {0, 0, 0, 0};
#if defined(__clang_major__) && (__clang_major__ >= 23)
  v8i_ z8 = {0, 0, 0, 0, 0, 0, 0, 0};
  __builtin_amdgcn_tensor_load_to_lds(g0, g1, z4, z4, z8, 0);
#else
  __builtin_amdgcn_tensor_load_to_lds(g0, g1, z4, z4, 0);
#endif
}

__global__ void k_gemm_tdm(const unsigned* __restrict__ Ap, const unsigned* __restrict__ Bp,
                           float* __restrict__ C, int Kt, int ldc,
                           const float* __restrict__ bias, int bias_mode,
                           int act, const float* __restrict__ resid) {
  extern __shared__ v16h smem[];                     // [4*Kt*32] A frags | [Kt*32] B frags
  int nt = blockIdx.x, mtBase = blockIdx.y << 2;
  int wave = threadIdx.x >> 5, lane = threadIdx.x & 31;
  unsigned aDwords = (unsigned)(Kt << 10);           // 4 strips * Kt * 256 dwords
  unsigned bDwords = (unsigned)(Kt << 8);
  if (wave == 0) {
    tdm_load_1d(Ap + ((size_t)mtBase * Kt << 8), 0u, aDwords);
    tdm_load_1d(Bp + ((size_t)nt * Kt << 8), aDwords << 2, bDwords);
    __builtin_amdgcn_s_wait_tensorcnt(0);
  }
  __syncthreads();
  const v16h* aS = smem + (size_t)wave * Kt * 32 + lane;
  const v16h* bS = smem + (size_t)4 * Kt * 32 + lane;
  v8f c0 = {}; v8f c1 = {};
  int pairs = Kt >> 1;
  for (int i = 0; i < pairs; ++i) {
    v16h a0 = aS[0],  b0 = bS[0];
    v16h a1 = aS[32], b1 = bS[32];
    c0 = __builtin_amdgcn_wmma_f32_16x16x32_f16(false, a0, false, b0, (short)0, c0, false, false);
    c1 = __builtin_amdgcn_wmma_f32_16x16x32_f16(false, a1, false, b1, (short)0, c1, false, false);
    aS += 64; bS += 64;
  }
  if (Kt & 1) {
    v16h a0 = aS[0], b0 = bS[0];
    c0 = __builtin_amdgcn_wmma_f32_16x16x32_f16(false, a0, false, b0, (short)0, c0, false, false);
  }
  gemm_epilogue(c0 + c1, C, mtBase + wave, nt, lane, ldc, bias, bias_mode, act, resid);
}

// ============================ BN1 + channel stats ====================
__global__ void k_bn_stats(const float* __restrict__ x, const float* g, const float* b,
                           const float* m, const float* v,
                           float* __restrict__ h, float* avg, float* mx) {
  int c = blockIdx.x, t = threadIdx.x;
  float sc = g[c] * rsqrtf(v[c] + 1e-5f);
  float sh = b[c] - m[c] * sc;
  const float* xi = x + (size_t)c * HW;
  float* ho = h + (size_t)c * HW;
  float s = 0.f, mv = -1e30f;
  for (int i = t; i < HW; i += 256) {
    float val = xi[i] * sc + sh;
    ho[i] = val; s += val; mv = fmaxf(mv, val);
  }
  __shared__ float ss[256], sm[256];
  ss[t] = s; sm[t] = mv; __syncthreads();
  for (int o = 128; o > 0; o >>= 1) {
    if (t < o) { ss[t] += ss[t + o]; sm[t] = fmaxf(sm[t], sm[t + o]); }
    __syncthreads();
  }
  if (t == 0) { avg[c] = ss[0] / (float)HW; mx[c] = sm[0]; }
}

// channel attention MLP (96 -> 6 -> 96), sigmoid(mlp(avg)+mlp(max))
__global__ void k_ca(const float* avg, const float* mx, const float* w1,
                     const float* w2, float* attn) {
  __shared__ float ha[8], hm[8];
  int t = threadIdx.x;
  if (t < 6) {
    float s = 0.f;
    for (int c = 0; c < CC; ++c) s += w1[t * CC + c] * avg[c];
    ha[t] = fmaxf(s, 0.f);
  } else if (t < 12) {
    int j = t - 6; float s = 0.f;
    for (int c = 0; c < CC; ++c) s += w1[j * CC + c] * mx[c];
    hm[j] = fmaxf(s, 0.f);
  }
  __syncthreads();
  if (t < CC) {
    float s = 0.f;
    for (int j = 0; j < 6; ++j) s += w2[t * 6 + j] * (ha[j] + hm[j]);
    attn[t] = 1.f / (1.f + __expf(-s));
  }
}

// xp[l][c*16+ph*4+pw] = h[c][(hp*4+ph)*128 + wp*4+pw], l = hp*32+wp
__global__ void k_patch_gather(const float* __restrict__ h, float* __restrict__ xp) {
  int tid = blockIdx.x * 256 + threadIdx.x;     // 1024*1536
  int l = tid / CPP, j = tid - l * CPP;
  int c = j >> 4, ph = (j >> 2) & 3, pw = j & 3;
  int hp = l >> 5, wp = l & 31;
  xp[tid] = h[(size_t)c * HW + (hp * 4 + ph) * WW_ + wp * 4 + pw];
}

// generic row layernorm; optional *silu(z) (z strided ldz)
__global__ void k_layernorm(const float* __restrict__ x, int ld, int W,
                            const float* g, const float* b,
                            float* __restrict__ out, int ldo,
                            const float* __restrict__ z, int ldz) {
  int row = blockIdx.x, t = threadIdx.x;
  const float* xr = x + (size_t)row * ld;
  float s = 0.f, s2 = 0.f;
  for (int i = t; i < W; i += 256) { float v = xr[i]; s += v; s2 += v * v; }
  __shared__ float ss[256], q[256];
  ss[t] = s; q[t] = s2; __syncthreads();
  for (int o = 128; o > 0; o >>= 1) {
    if (t < o) { ss[t] += ss[t + o]; q[t] += q[t + o]; }
    __syncthreads();
  }
  float mu  = ss[0] / (float)W;
  float var = q[0] / (float)W - mu * mu;
  float inv = rsqrtf(var + 1e-5f);
  for (int i = t; i < W; i += 256) {
    float v = (xr[i] - mu) * inv * g[i] + b[i];
    if (z) { float zz = z[(size_t)row * ldz + i]; v *= zz / (1.f + __expf(-zz)); }
    out[(size_t)row * ldo + i] = v;
  }
}

// depthwise 3x3 conv over xz[:, :DI] (NHWC rows of stride 768) + bias + silu
__global__ void k_dwconv(const float* __restrict__ xz, const float* __restrict__ w9,
                         const float* __restrict__ bias, float* __restrict__ xc) {
  int tid = blockIdx.x * 256 + threadIdx.x;     // 384*1024
  int d = tid >> 10, l = tid & 1023;
  int hh = l >> 5, ww = l & 31;
  const float* wd = w9 + (size_t)d * 9;
  float s = bias[d];
#pragma unroll
  for (int ky = 0; ky < 3; ++ky) {
    int y = hh + ky - 1;
    if (y < 0 || y >= HP) continue;
#pragma unroll
    for (int kx = 0; kx < 3; ++kx) {
      int xw = ww + kx - 1;
      if (xw < 0 || xw >= HP) continue;
      s += wd[ky * 3 + kx] * xz[(size_t)(y * HP + xw) * 768 + d];
    }
  }
  xc[tid] = s / (1.f + __expf(-s));             // silu
}

// build 4 scan directions: xs[k][d][l]
__global__ void k_build_xs(const float* __restrict__ xc, float* __restrict__ xs) {
  int tid = blockIdx.x * 256 + threadIdx.x;     // 384*1024
  int d = tid >> 10, l = tid & 1023;
  int hh = l >> 5, ww = l & 31;
  float v = xc[tid];
  int lt = ww * HP + hh;                        // (h,w) -> transposed slot
  xs[((size_t)0 * DI + d) * LL + l] = v;
  xs[((size_t)1 * DI + d) * LL + lt] = v;
  xs[((size_t)2 * DI + d) * LL + (LL - 1 - l)] = v;
  xs[((size_t)3 * DI + d) * LL + (LL - 1 - lt)] = v;
}

// selective scan: thread = (k,d), 16 states in registers, serial over L
__global__ void k_scan(const float* __restrict__ xs, const float* __restrict__ dts,
                       const float* __restrict__ xdbl, const float* __restrict__ Alog,
                       const float* __restrict__ Dsk, float* __restrict__ oy) {
  int k = blockIdx.x;
  int d = blockIdx.y * 128 + threadIdx.x;
  const float* xrow  = xs  + ((size_t)k * DI + d) * LL;
  const float* dtrow = dts + ((size_t)k * DI + d) * LL;
  const float* Brow  = xdbl + ((size_t)k * 48 + RRk)          * LL;
  const float* Crow  = xdbl + ((size_t)k * 48 + RRk + NSZ)    * LL;
  float A[NSZ], hst[NSZ];
#pragma unroll
  for (int n = 0; n < NSZ; ++n) {
    A[n] = -__expf(Alog[((size_t)k * DI + d) * NSZ + n]);
    hst[n] = 0.f;
  }
  float Dv = Dsk[k * DI + d];
  float* orow = oy + ((size_t)k * DI + d) * LL;
  for (int l = 0; l < LL; ++l) {
    float xv = xrow[l], dt = dtrow[l];
    float dx = dt * xv, y = 0.f;
#pragma unroll
    for (int n = 0; n < NSZ; ++n) {
      float e = __expf(dt * A[n]);
      hst[n] = e * hst[n] + dx * Brow[(size_t)n * LL + l];
      y += hst[n] * Crow[(size_t)n * LL + l];
    }
    orow[l] = y + xv * Dv;
  }
}

// merge 4 directions (undo transposes/flips) -> y[l][d]
__global__ void k_merge(const float* __restrict__ oy, const float* __restrict__ mw,
                        float* __restrict__ y) {
  int tid = blockIdx.x * 256 + threadIdx.x;     // 1024*384
  int l = tid / DI, d = tid - l * DI;
  int hh = l >> 5, ww = l & 31;
  int lt = ww * HP + hh;
  float v = mw[0] * oy[((size_t)0 * DI + d) * LL + l]
          + mw[1] * oy[((size_t)1 * DI + d) * LL + lt]
          + mw[2] * oy[((size_t)2 * DI + d) * LL + (LL - 1 - l)]
          + mw[3] * oy[((size_t)3 * DI + d) * LL + (LL - 1 - lt)];
  y[(size_t)l * DI + d] = v;
}

// xsum = x + h*attn[c] + unpatch(xu)
__global__ void k_xsum(const float* __restrict__ x, const float* __restrict__ h,
                       const float* __restrict__ attn, const float* __restrict__ xu,
                       float* __restrict__ xsum) {
  int tid = blockIdx.x * 256 + threadIdx.x;     // 96*16384
  int c = tid >> 14, i = tid & (HW - 1);
  int hh = i >> 7, ww = i & 127;
  int l = (hh >> 2) * HP + (ww >> 2);
  int j = c * 16 + (hh & 3) * 4 + (ww & 3);
  xsum[tid] = x[tid] + h[tid] * attn[c] + xu[(size_t)l * CPP + j];
}

__global__ void k_bn(const float* __restrict__ x, const float* g, const float* b,
                     const float* m, const float* v, float* __restrict__ out) {
  int tid = blockIdx.x * 256 + threadIdx.x;     // 96*16384
  int c = tid >> 14;
  float sc = g[c] * rsqrtf(v[c] + 1e-5f);
  out[tid] = (x[tid] - m[c]) * sc + b[c];
}

// direct 3x3 conv, 48 in/out channels, 128x128; optional *mul, optional lrelu
__global__ void k_conv3(const float* __restrict__ in, const float* __restrict__ w,
                        float* __restrict__ out, const float* __restrict__ mul, int act) {
  int tid = blockIdx.x * 256 + threadIdx.x;     // 48*16384
  int o = tid >> 14, i = tid & (HW - 1);
  int y = i >> 7, x = i & 127;
  float s = 0.f;
  for (int c = 0; c < GWC; ++c) {
    const float* wp = w + ((size_t)o * GWC + c) * 9;
    const float* ip = in + (size_t)c * HW;
#pragma unroll
    for (int ky = 0; ky < 3; ++ky) {
      int yy = y + ky - 1;
      if (yy < 0 || yy >= HH) continue;
#pragma unroll
      for (int kx = 0; kx < 3; ++kx) {
        int xx = x + kx - 1;
        if (xx < 0 || xx >= WW_) continue;
        s += wp[ky * 3 + kx] * ip[yy * WW_ + xx];
      }
    }
  }
  if (mul) s *= mul[tid];
  if (act == 1) s = s >= 0.f ? s : 0.2f * s;
  out[tid] = s;
}

__global__ void k_cat_lrelu(const float* __restrict__ a, const float* __restrict__ b,
                            float* __restrict__ cat) {
  int tid = blockIdx.x * 256 + threadIdx.x;     // 48*16384
  float av = a[tid], bv = b[tid];
  cat[tid]                     = av >= 0.f ? av : 0.2f * av;
  cat[(size_t)GWC * HW + tid]  = bv >= 0.f ? bv : 0.2f * bv;
}

__global__ void k_final(const float* __restrict__ c3o, const float* __restrict__ h2,
                        const float* __restrict__ xsum, float* __restrict__ out) {
  int tid = blockIdx.x * 256 + threadIdx.x;     // 96*16384
  out[tid] = c3o[tid] + h2[tid] + xsum[tid];
}

// =====================================================================
extern "C" void kernel_launch(void* const* d_in, const int* in_sizes, int n_in,
                              void* d_out, int out_size, void* d_ws, size_t ws_size,
                              hipStream_t stream) {
  (void)in_sizes; (void)n_in; (void)out_size; (void)ws_size;
  const float* x          = (const float*)d_in[0];
  const float* bn1_g      = (const float*)d_in[1];
  const float* bn1_b      = (const float*)d_in[2];
  const float* bn1_m      = (const float*)d_in[3];
  const float* bn1_v      = (const float*)d_in[4];
  const float* ca_w1      = (const float*)d_in[5];
  const float* ca_w2      = (const float*)d_in[6];
  const float* pe_w       = (const float*)d_in[7];
  const float* pe_b       = (const float*)d_in[8];
  const float* pu_w       = (const float*)d_in[9];
  const float* pu_b       = (const float*)d_in[10];
  const float* bln_g      = (const float*)d_in[11];
  const float* bln_b      = (const float*)d_in[12];
  const float* binproj_w  = (const float*)d_in[13];
  const float* bconv_w    = (const float*)d_in[14];
  const float* bconv_b    = (const float*)d_in[15];
  const float* bxproj_w   = (const float*)d_in[16];
  const float* bdtproj_w  = (const float*)d_in[17];
  const float* bdtproj_b  = (const float*)d_in[18];
  const float* bAlog      = (const float*)d_in[19];
  const float* bD         = (const float*)d_in[20];
  const float* bon_g      = (const float*)d_in[21];
  const float* bon_b      = (const float*)d_in[22];
  const float* boutproj_w = (const float*)d_in[23];
  const float* bmerge_w   = (const float*)d_in[24];
  const float* bn2_g      = (const float*)d_in[25];
  const float* bn2_b      = (const float*)d_in[26];
  const float* bn2_m      = (const float*)d_in[27];
  const float* bn2_v      = (const float*)d_in[28];
  const float* c1a_w      = (const float*)d_in[29];
  const float* c1b_w      = (const float*)d_in[30];
  const float* k1_w       = (const float*)d_in[31];
  const float* k2_w       = (const float*)d_in[32];
  const float* k2_b       = (const float*)d_in[33];
  const float* k3_w       = (const float*)d_in[34];
  const float* k4_w       = (const float*)d_in[35];
  const float* c3_w       = (const float*)d_in[36];
  float* out = (float*)d_out;
  char* ws = (char*)d_ws;

  // deterministic bump allocator (~54 MB, with region reuse)
  size_t off = 0;
  auto alloc = [&](size_t bytes) { size_t o = off; off += (bytes + 255) & ~(size_t)255; return o; };
  const size_t oH    = alloc((size_t)CC * HW * 4);        // h (bn1 out); later h2
  const size_t oST   = alloc(4096);                       // avg/max/attn
  const size_t oX2   = alloc((size_t)LL * EE * 4);        // token stream (updated in place)
  const size_t oLNX  = alloc((size_t)LL * DI * 4);        // layernorm outputs
  const size_t oAPK  = alloc((size_t)LL * CPP * 2);       // packed A fragments (max 3.15MB)
  const size_t oBPK  = alloc((size_t)1024 * 3 * 256 * 4); // packed B fragments (max 3.15MB)
  const size_t oXZ   = alloc((size_t)LL * 768 * 4);       // xz; later scpa 'a', then 'b2'
  const size_t oXC   = alloc((size_t)DI * LL * 4);        // dwconv out
  const size_t oXS   = alloc((size_t)KDIR * DI * LL * 4); // xs; later scpa 'b'
  const size_t oXDBL = alloc((size_t)KDIR * 48 * LL * 4); // x_dbl (44 padded to 48 rows)
  const size_t oDTS  = alloc((size_t)KDIR * DI * LL * 4); // dts; later 'g', then c3out
  const size_t oOY   = alloc((size_t)KDIR * DI * LL * 4); // scan out; later a1conv
  const size_t oY    = alloc((size_t)LL * DI * 4);        // merged y
  const size_t oXPU  = alloc((size_t)LL * CPP * 4);       // xp; later xu, then t, then cat
  const size_t oXSUM = alloc((size_t)CC * HW * 4);

  float* H    = (float*)(ws + oH);
  float* Avg  = (float*)(ws + oST);
  float* Mx   = Avg + 128;
  float* Attn = Avg + 256;
  float* X2   = (float*)(ws + oX2);
  float* LNX  = (float*)(ws + oLNX);
  unsigned* APK = (unsigned*)(ws + oAPK);
  unsigned* BPK = (unsigned*)(ws + oBPK);
  float* XZ   = (float*)(ws + oXZ);
  float* XC   = (float*)(ws + oXC);
  float* XS   = (float*)(ws + oXS);
  float* XDBL = (float*)(ws + oXDBL);
  float* DTS  = (float*)(ws + oDTS);
  float* OY   = (float*)(ws + oOY);
  float* Y    = (float*)(ws + oY);
  float* XPU  = (float*)(ws + oXPU);
  float* XSUM = (float*)(ws + oXSUM);

  auto packA = [&](const float* src, int M, int Kd, int ld, int Mt, int Kt) {
    k_pack_a<<<dim3(Mt * Kt), 256, 0, stream>>>(src, APK, M, Kd, ld, Kt);
  };
  auto packB = [&](const float* src, int Kd, int N, int ld, int Nt, int Kt, int trans) {
    k_pack_b<<<dim3(Nt * Kt), 256, 0, stream>>>(src, BPK, Kd, N, ld, Kt, trans);
  };
  // TDM-staged path when a block of 4 M-tiles is available (LDS = 5*Kt KB <= 240KB)
  auto gemm = [&](float* C, int Mt, int Nt, int Kt, int ldc,
                  const float* bias, int bmode, int act, const float* resid) {
    if ((Mt & 3) == 0) {
      size_t shbytes = (size_t)5 * Kt * 1024;
      k_gemm_tdm<<<dim3(Nt, Mt >> 2), 128, shbytes, stream>>>(APK, BPK, C, Kt, ldc,
                                                              bias, bmode, act, resid);
    } else {
      k_gemm<<<dim3(Nt, Mt), 32, 0, stream>>>((const v16h*)APK, (const v16h*)BPK,
                                              C, Kt, ldc, bias, bmode, act, resid);
    }
  };

  // ---------- stage 1: bn1 + channel attention stats ----------
  k_bn_stats<<<dim3(CC), 256, 0, stream>>>(x, bn1_g, bn1_b, bn1_m, bn1_v, H, Avg, Mx);
  k_ca<<<dim3(1), 128, 0, stream>>>(Avg, Mx, ca_w1, ca_w2, Attn);

  // ---------- stage 2: patch embed (WMMA GEMM 1024x1536x192) ----------
  k_patch_gather<<<dim3(LL * CPP / 256), 256, 0, stream>>>(H, XPU);
  packA(XPU, LL, CPP, CPP, 64, 48);
  packB(pe_w, CPP, EE, CPP, 12, 48, 1);
  gemm(X2, 64, 12, 48, EE, pe_b, 1, 0, nullptr);

  // ---------- stage 3: 8 SS2D blocks ----------
  for (int blk = 0; blk < NBLK; ++blk) {
    // layernorm + in_proj GEMM (1024x192x768)
    k_layernorm<<<dim3(LL), 256, 0, stream>>>(X2, EE, EE, bln_g + blk * EE, bln_b + blk * EE,
                                              LNX, EE, nullptr, 0);
    packA(LNX, LL, EE, EE, 64, 6);
    packB(binproj_w + (size_t)blk * 768 * EE, EE, 768, EE, 48, 6, 1);
    gemm(XZ, 64, 48, 6, 768, nullptr, 0, 0, nullptr);

    // depthwise conv + silu, build 4 scan directions
    k_dwconv<<<dim3(DI * LL / 256), 256, 0, stream>>>(XZ, bconv_w + (size_t)blk * DI * 9,
                                                      bconv_b + blk * DI, XC);
    k_build_xs<<<dim3(DI * LL / 256), 256, 0, stream>>>(XC, XS);

    // x_proj per direction: [48pad x 1024] = W[44x384] * xs[384x1024]
    for (int k = 0; k < KDIR; ++k) {
      packA(bxproj_w + ((size_t)blk * KDIR + k) * 44 * DI, 44, DI, DI, 3, 12);
      packB(XS + (size_t)k * DI * LL, DI, LL, LL, 64, 12, 0);
      gemm(XDBL + (size_t)k * 48 * LL, 3, 64, 12, LL, nullptr, 0, 0, nullptr);
    }
    // dt_proj per direction + softplus epilogue: [384x1024] = W[384x12] * dts[12x1024]
    for (int k = 0; k < KDIR; ++k) {
      packA(bdtproj_w + ((size_t)blk * KDIR + k) * DI * RRk, DI, RRk, RRk, 24, 1);
      packB(XDBL + (size_t)k * 48 * LL, RRk, LL, LL, 64, 1, 0);
      gemm(DTS + (size_t)k * DI * LL, 24, 64, 1, LL,
           bdtproj_b + ((size_t)blk * KDIR + k) * DI, 2, 3, nullptr);
    }
    // selective scan (register-resident 16-state recurrence)
    k_scan<<<dim3(KDIR, DI / 128), 128, 0, stream>>>(XS, DTS, XDBL,
        bAlog + (size_t)blk * KDIR * DI * NSZ, bD + (size_t)blk * KDIR * DI, OY);

    // merge directions, out-norm * silu(z), out_proj (+residual into X2)
    k_merge<<<dim3(LL * DI / 256), 256, 0, stream>>>(OY, bmerge_w + blk * KDIR, Y);
    k_layernorm<<<dim3(LL), 256, 0, stream>>>(Y, DI, DI, bon_g + blk * DI, bon_b + blk * DI,
                                              LNX, DI, XZ + DI, 768);
    packA(LNX, LL, DI, DI, 64, 12);
    packB(boutproj_w + (size_t)blk * EE * DI, DI, EE, DI, 12, 12, 1);
    gemm(X2, 64, 12, 12, EE, nullptr, 0, 0, X2);
  }

  // ---------- stage 4: patch unembed + residual sum + bn2 ----------
  packA(X2, LL, EE, EE, 64, 6);
  packB(pu_w, EE, CPP, EE, 96, 6, 1);
  gemm(XPU, 64, 96, 6, CPP, pu_b, 1, 0, nullptr);
  k_xsum<<<dim3(CC * HW / 256), 256, 0, stream>>>(x, H, Attn, XPU, XSUM);
  k_bn<<<dim3(CC * HW / 256), 256, 0, stream>>>(XSUM, bn2_g, bn2_b, bn2_m, bn2_v, H); // H = h2

  // ---------- stage 5: SCPA ----------
  // 1x1 convs via WMMA (A = weights, B = h2 [96 x 16384])
  packB(H, CC, HW, HW, 1024, 3, 0);
  packA(c1a_w, GWC, CC, CC, 3, 3);
  gemm(XZ, 3, 1024, 3, HW, nullptr, 0, 1, nullptr);        // a = lrelu(c1a(h2))
  packA(c1b_w, GWC, CC, CC, 3, 3);
  gemm(XS, 3, 1024, 3, HW, nullptr, 0, 1, nullptr);        // b = lrelu(c1b(h2))

  k_conv3<<<dim3(GWC * HW / 256), 256, 0, stream>>>(XZ, k1_w, OY, nullptr, 0);   // a1 = conv(a,k1)

  packA(k2_w, GWC, GWC, GWC, 3, 2);
  packB(XS, GWC, HW, HW, 1024, 2, 0);
  gemm(DTS, 3, 1024, 2, HW, k2_b, 2, 2, nullptr);          // g = sigmoid(k2(b)+bias)

  k_conv3<<<dim3(GWC * HW / 256), 256, 0, stream>>>(XS, k3_w, XPU, DTS, 0);      // t = conv(b,k3)*g
  k_conv3<<<dim3(GWC * HW / 256), 256, 0, stream>>>(XPU, k4_w, XZ, nullptr, 0);  // b2 = conv(t,k4)

  k_cat_lrelu<<<dim3(GWC * HW / 256), 256, 0, stream>>>(OY, XZ, XPU);            // cat [96 x 16384]

  packA(c3_w, CC, 2 * GWC, 2 * GWC, 6, 3);
  packB(XPU, 2 * GWC, HW, HW, 1024, 3, 0);
  gemm(DTS, 6, 1024, 3, HW, nullptr, 0, 0, nullptr);       // c3out

  k_final<<<dim3(CC * HW / 256), 256, 0, stream>>>(DTS, H, XSUM, out);
}
